// SimVQ_1657857376701
// MI455X (gfx1250) — matrix operand order repaired and checked by reading, not compiled, not run on hardware
//
#include <hip/hip_runtime.h>
#include <math.h>

// SimVQ on MI455X (gfx1250, wave32, WMMA).
// Fused z@C^T GEMM + argmin via v_wmma_f32_16x16x32_bf16 with bf16 hi/lo
// (bf16x3) split for ~fp32 distances. B tiles staged once per block in LDS
// (double-buffered, bank-conflict-free padding), 2 M-tiles per wave.

#define T_TOKENS 32768   // 8*4096
#define KCB      8192
#define D        128
#define NTILES   (KCB / 16)          // 512 N-tiles of 16 codes
#define LDS_ROW4 17                  // row stride in float4 units (272 B)
#define LDS_BUF4 (32 * LDS_ROW4)     // 32 rows (16 hi + 16 lo) per buffer

typedef __attribute__((ext_vector_type(16))) __bf16 v16bf;
typedef __attribute__((ext_vector_type(8)))  float  v8f;

__device__ __forceinline__ void split_bf16(float x, __bf16& hi, __bf16& lo) {
  hi = (__bf16)x;                 // RNE convert
  lo = (__bf16)(x - (float)hi);   // residual
}

// ---------------------------------------------------------------------------
// Kernel 0: zero the histogram (harness poisons workspace with 0xAA).
__global__ void zero_ws(unsigned int* __restrict__ counts) {
  int i = blockIdx.x * blockDim.x + threadIdx.x;
  if (i < KCB) counts[i] = 0u;
}

// ---------------------------------------------------------------------------
// Kernel 1: C[k][d] = dot(emb[k,:], W[d,:]) + b[d];  bf16 hi/lo split and
// cnorm[k] = |C[k,:]|^2.  One block (128 threads) per codebook row.
__global__ void project_codebook(const float* __restrict__ emb,
                                 const float* __restrict__ W,
                                 const float* __restrict__ b,
                                 float*  __restrict__ C,
                                 __bf16* __restrict__ Chi,
                                 __bf16* __restrict__ Clo,
                                 float*  __restrict__ cnorm) {
  const int k = blockIdx.x;
  const int d = threadIdx.x;     // 0..127
  __shared__ float erow[D];
  __shared__ float red[D];
  erow[d] = emb[k * D + d];
  __syncthreads();
  float acc = b[d];
  const float* wr = W + d * D;
  #pragma unroll 8
  for (int j = 0; j < D; ++j) acc = fmaf(erow[j], wr[j], acc);
  C[k * D + d] = acc;
  __bf16 hi, lo;
  split_bf16(acc, hi, lo);
  Chi[k * D + d] = hi;
  Clo[k * D + d] = lo;
  red[d] = acc * acc;
  __syncthreads();
  for (int s = 64; s > 0; s >>= 1) {
    if (d < s) red[d] += red[d + s];
    __syncthreads();
  }
  if (d == 0) cnorm[k] = red[0];
}

// ---------------------------------------------------------------------------
// Kernel 2: fused GEMM + argmin.
// Block = 256 threads = 8 waves; each wave owns TWO 16-row M-tiles (block
// covers 256 rows).  B tiles (16 codes x 128 dims, hi+lo bf16) are staged
// cooperatively in double-buffered LDS; rows padded to 272 B so half-wave
// fragment reads are bank-conflict-free.  Per staged tile each wave issues
// 4 K-chunks x 3 (hh,hl,lh) x 2 M-tiles = 24 WMMAs.
__global__ __launch_bounds__(256) void
vq_argmin(const float*  __restrict__ z,
          const __bf16* __restrict__ Chi,
          const __bf16* __restrict__ Clo,
          const float*  __restrict__ cnorm,
          int* __restrict__ idx) {
  const int tid  = threadIdx.x;
  const int lane = tid & 31;
  const int wave = tid >> 5;
  const int half = lane >> 4;        // 0: lanes 0-15, 1: lanes 16-31
  const int l15  = lane & 15;
  const int m0   = blockIdx.x * 256 + wave * 32;

  __shared__ float4 smem[2 * LDS_BUF4];   // 2 x 8704 B

  // ---- A fragments for two 16-row tiles (contiguous K order, same as B) ----
  //   elems 0..7  -> K = d0 + half*16 + {0..7}
  //   elems 8..15 -> K = d0 + half*16 + 8 + {0..7}
  v16bf a_hi[2][4], a_lo[2][4];
  #pragma unroll
  for (int t = 0; t < 2; ++t) {
    const int mrow = m0 + t * 16 + l15;
    #pragma unroll
    for (int kc = 0; kc < 4; ++kc) {
      const float* zp = z + (size_t)mrow * D + kc * 32 + half * 16;
      #pragma unroll
      for (int j = 0; j < 8; ++j) {
        __bf16 h, l;
        split_bf16(zp[j], h, l);
        a_hi[t][kc][j] = h; a_lo[t][kc][j] = l;
        __bf16 h2, l2;
        split_bf16(zp[8 + j], h2, l2);
        a_hi[t][kc][8 + j] = h2; a_lo[t][kc][8 + j] = l2;
      }
    }
  }

  // ---- cooperative stage: thread -> (row = tid>>4, seg = tid&15) ----
  const int srow = tid >> 4;   // 0..15  (code row within tile)
  const int sseg = tid & 15;   // 16B segment within the 256B row
  const float4* gsrc_hi0 = reinterpret_cast<const float4*>(Chi) + (size_t)srow * 8 + sseg;
  const float4* gsrc_lo0 = reinterpret_cast<const float4*>(Clo) + (size_t)srow * 8 + sseg;
  // each N-tile advances the source by 128 float4 = 8 rows * 128 bf16
  float4* ldst_hi0 = smem + srow * LDS_ROW4 + sseg;
  float4* ldst_lo0 = smem + (16 + srow) * LDS_ROW4 + sseg;

  // prologue: fill buffer 0 with tile 0
  {
    float4 ph = gsrc_hi0[0];
    float4 pl = gsrc_lo0[0];
    ldst_hi0[0] = ph;
    ldst_lo0[0] = pl;
  }
  __syncthreads();

  float minv[2][8];
  int   mini[2][8];
  #pragma unroll
  for (int t = 0; t < 2; ++t)
    #pragma unroll
    for (int v = 0; v < 8; ++v) { minv[t][v] = 3.4e38f; mini[t][v] = 0; }

  for (int nt = 0; nt < NTILES; ++nt) {
    const int cur = nt & 1;
    const bool more = (nt + 1 < NTILES);
    // prefetch next tile into registers while computing this one
    float4 ph, pl;
    if (more) {
      ph = gsrc_hi0[(size_t)(nt + 1) * 128];
      pl = gsrc_lo0[(size_t)(nt + 1) * 128];
    }

    const int n = nt * 16 + l15;   // this lane's codebook column
    v8f acc0 = {}, acc1 = {};
    const float4* bufp = smem + cur * LDS_BUF4;
    #pragma unroll
    for (int kc = 0; kc < 4; ++kc) {
      // B fragment (32x16 bf16): 16 contiguous bf16 per lane, K-half by
      // lane group; read from LDS (ds_load_b128 x2, conflict-free).
      const v16bf bh = *reinterpret_cast<const v16bf*>(
          bufp + l15 * LDS_ROW4 + kc * 4 + half * 2);
      const v16bf bl = *reinterpret_cast<const v16bf*>(
          bufp + (16 + l15) * LDS_ROW4 + kc * 4 + half * 2);
      acc0 = __builtin_amdgcn_wmma_f32_16x16x32_bf16(false, a_hi[0][kc], false, bh,
                                                     (short)0, acc0, false, false);
      acc0 = __builtin_amdgcn_wmma_f32_16x16x32_bf16(false, a_hi[0][kc], false, bl,
                                                     (short)0, acc0, false, false);
      acc0 = __builtin_amdgcn_wmma_f32_16x16x32_bf16(false, a_lo[0][kc], false, bh,
                                                     (short)0, acc0, false, false);
      acc1 = __builtin_amdgcn_wmma_f32_16x16x32_bf16(false, a_hi[1][kc], false, bh,
                                                     (short)0, acc1, false, false);
      acc1 = __builtin_amdgcn_wmma_f32_16x16x32_bf16(false, a_hi[1][kc], false, bl,
                                                     (short)0, acc1, false, false);
      acc1 = __builtin_amdgcn_wmma_f32_16x16x32_bf16(false, a_lo[1][kc], false, bh,
                                                     (short)0, acc1, false, false);
    }
    const float cn = cnorm[n];
    // C/D layout: VGPR v -> row M = v + half*8, col N = n (per lane).
    #pragma unroll
    for (int v = 0; v < 8; ++v) {
      float d0 = cn - 2.0f * acc0[v];
      if (d0 < minv[0][v]) { minv[0][v] = d0; mini[0][v] = n; }
      float d1 = cn - 2.0f * acc1[v];
      if (d1 < minv[1][v]) { minv[1][v] = d1; mini[1][v] = n; }
    }

    __syncthreads();               // all waves done reading buf[1-cur] long ago,
    if (more) {                    // and done reading buf[cur] just now
      ldst_hi0[(cur ^ 1) * LDS_BUF4] = ph;
      ldst_lo0[(cur ^ 1) * LDS_BUF4] = pl;
    }
    __syncthreads();               // fill visible before next compute
  }

  // Argmin across the 16 lanes holding each row (xor butterfly, tie -> lower n)
  #pragma unroll
  for (int t = 0; t < 2; ++t) {
    #pragma unroll
    for (int v = 0; v < 8; ++v) {
      float mv = minv[t][v];
      int   mi = mini[t][v];
      #pragma unroll
      for (int m = 1; m < 16; m <<= 1) {
        float ov = __shfl_xor(mv, m, 32);
        int   oi = __shfl_xor(mi, m, 32);
        if (ov < mv || (ov == mv && oi < mi)) { mv = ov; mi = oi; }
      }
      if (l15 == 0) idx[m0 + t * 16 + half * 8 + v] = mi;
    }
  }
}

// ---------------------------------------------------------------------------
// Kernel 3: gather z_q (== z_q_st numerically), per-token squared-error
// partial (deterministic), histogram count via integer atomics.
__global__ void gather_stats(const float* __restrict__ z,
                             const float* __restrict__ C,
                             const int*   __restrict__ idx,
                             float* __restrict__ out,
                             unsigned int* __restrict__ counts,
                             float* __restrict__ partial) {
  const int t = blockIdx.x;
  const int d = threadIdx.x;
  const int id = idx[t];
  float q    = C[(size_t)id * D + d];
  float diff = q - z[(size_t)t * D + d];
  out[(size_t)t * D + d] = q;                 // z + sg(z_q - z) == z_q
  __shared__ float red[D];
  red[d] = diff * diff;
  __syncthreads();
  for (int s = 64; s > 0; s >>= 1) {
    if (d < s) red[d] += red[d + s];
    __syncthreads();
  }
  if (d == 0) {
    partial[t] = red[0];
    atomicAdd(&counts[id], 1u);
  }
}

// ---------------------------------------------------------------------------
// Kernel 4: fixed-order reductions -> commit_loss, perplexity.
__global__ void finalize(const unsigned int* __restrict__ counts,
                         const float* __restrict__ partial,
                         float* __restrict__ out) {
  const int tid = threadIdx.x;   // 256 threads
  __shared__ float red[256];
  float s = 0.0f;
  for (int i = tid; i < T_TOKENS; i += 256) s += partial[i];
  red[tid] = s;
  __syncthreads();
  for (int st = 128; st > 0; st >>= 1) {
    if (tid < st) red[tid] += red[tid + st];
    __syncthreads();
  }
  const float sqsum = red[0];
  __syncthreads();
  float h = 0.0f;
  for (int i = tid; i < KCB; i += 256) {
    float e = (float)counts[i] / (float)T_TOKENS;
    h += e * logf(e + 1e-8f);
  }
  red[tid] = h;
  __syncthreads();
  for (int st = 128; st > 0; st >>= 1) {
    if (tid < st) red[tid] += red[tid + st];
    __syncthreads();
  }
  if (tid == 0) {
    // beta*mean + mean == 1.25 * MSE numerically (stop_gradient is identity)
    out[(size_t)T_TOKENS * D]     = 1.25f * sqsum / (float)((size_t)T_TOKENS * D);
    out[(size_t)T_TOKENS * D + 1] = expf(-red[0]);
  }
}

// ---------------------------------------------------------------------------
extern "C" void kernel_launch(void* const* d_in, const int* in_sizes, int n_in,
                              void* d_out, int out_size, void* d_ws, size_t ws_size,
                              hipStream_t stream) {
  const float* z   = (const float*)d_in[0];  // [8,4096,128]
  const float* emb = (const float*)d_in[1];  // [8192,128]
  const float* W   = (const float*)d_in[2];  // [128,128]
  const float* b   = (const float*)d_in[3];  // [128]
  float* out = (float*)d_out;                // z_q_st | commit | perplexity

  char* ws = (char*)d_ws;
  float*        C       = (float*)       (ws + 0);                        // 4 MB
  __bf16*       Chi     = (__bf16*)      (ws + 4u * 1024 * 1024);         // 2 MB
  __bf16*       Clo     = (__bf16*)      (ws + 6u * 1024 * 1024);         // 2 MB
  float*        cnorm   = (float*)       (ws + 8u * 1024 * 1024);         // 32 KB
  int*          idx     = (int*)         (ws + 8u * 1024 * 1024 + 64  * 1024); // 128 KB
  unsigned int* counts  = (unsigned int*)(ws + 8u * 1024 * 1024 + 256 * 1024); // 32 KB
  float*        partial = (float*)       (ws + 8u * 1024 * 1024 + 320 * 1024); // 128 KB

  zero_ws<<<(KCB + 255) / 256, 256, 0, stream>>>(counts);
  project_codebook<<<KCB, D, 0, stream>>>(emb, W, b, C, Chi, Clo, cnorm);
  vq_argmin<<<T_TOKENS / 256, 256, 0, stream>>>(z, Chi, Clo, cnorm, idx);
  gather_stats<<<T_TOKENS, D, 0, stream>>>(z, C, idx, out, counts, partial);
  finalize<<<1, 256, 0, stream>>>(counts, partial, out);
}
